// TransformerBlock_36232344109735
// MI455X (gfx1250) — matrix operand order repaired and checked by reading, not compiled, not run on hardware
//
#include <hip/hip_runtime.h>
#include <hip/hip_bf16.h>
#include <stdint.h>

// ---------------------------------------------------------------------------
// Transformer block for MI455X (gfx1250, wave32, WMMA).
// All GEMMs run through v_wmma_f32_16x16x32_f16 (f16 inputs, f32 accumulate),
// with double-buffered LDS staging, vectorized global loads, and all four
// B fragments resident in distinct registers so the 4 WMMAs per K-step issue
// back-to-back without hazard NOPs.
// Softmax / LayerNorm / residuals stay in fp32.
// ---------------------------------------------------------------------------

typedef __attribute__((ext_vector_type(16))) _Float16 v16h;
typedef __attribute__((ext_vector_type(8)))  float    v8f;
typedef __attribute__((ext_vector_type(4)))  _Float16 h4;

#define D_MODEL 1024
#define N_HEADS 16
#define DK      64
#define D_FF    4096
#define SEQ     2048
#define BSZ     2
#define LN_EPS  1e-5f

// GEMM tiling
#define BM 64
#define BN 64
#define BKK 32
#define LDA_S 36   // LDS row stride (halfs): keeps h4 stores 8B-aligned, spreads banks
#define LDB_S 36

__device__ inline _Float16 to_h(float x)    { return (_Float16)x; }
__device__ inline _Float16 to_h(_Float16 x) { return x; }

template<typename T> struct __align__(4 * sizeof(T)) V4 { T e[4]; };

union Frag { v16h v; unsigned u[8]; };

// ---------------------------------------------------------------------------
// Generic tiled WMMA GEMM:  C[z] = act(A[z] (MxK) * B[z] (KxN) + bias)
//   TA/TB: float (converted to f16 during staging) or _Float16
//   B_TRANS: if true, global B is indexed as B[n*ldb + k]  (i.e. A @ B^T)
//   RELU: apply max(0, .) in epilogue
//   OUT_HALF: write _Float16 to Ch, else float to Cf
// Batch index z (blockIdx.z) decomposes as (z/innerCnt, z%innerCnt) with
// independent outer/inner strides per operand.
// Requires M % 64 == 0, N % 64 == 0, K % 32 == 0 and all leading dims /
// operand offsets multiples of 4 elements (true for every call here).
// ---------------------------------------------------------------------------
template<typename TA, typename TB, bool B_TRANS, bool RELU, bool OUT_HALF>
__global__ __launch_bounds__(128)
void wmma_gemm_kernel(const TA* __restrict__ Ag, const TB* __restrict__ Bg,
                      const float* __restrict__ bias,
                      float* __restrict__ Cf, _Float16* __restrict__ Ch,
                      int M, int N, int K,
                      int lda, int ldb, int ldc,
                      int innerCnt,
                      long long sAo, long long sAi,
                      long long sBo, long long sBi,
                      long long sCo, long long sCi)
{
    __shared__ __attribute__((aligned(16))) _Float16 aT[2][BM * LDA_S];
    __shared__ __attribute__((aligned(16))) _Float16 bT[2][BN * LDB_S];

    const int z  = blockIdx.z;
    const int zo = z / innerCnt, zi = z % innerCnt;
    const TA* A = Ag + (long long)zo * sAo + (long long)zi * sAi;
    const TB* B = Bg + (long long)zo * sBo + (long long)zi * sBi;
    const long long offC = (long long)zo * sCo + (long long)zi * sCi;

    const int m0   = blockIdx.y * BM;
    const int n0   = blockIdx.x * BN;
    const int t    = threadIdx.x;     // 0..127
    const int wave = t >> 5;          // 0..3 : each wave owns 16 rows
    const int lane = t & 31;
    const int lr   = lane & 15;
    const int hi   = lane >> 4;

    // ---- precomputed staging addresses (loop-invariant) ------------------
    // A: 64 rows x 32 cols = 512 vec4; 4 per thread.
    const TA* aPtr[4]; int aOffS[4];
    #pragma unroll
    for (int i = 0; i < 4; ++i) {
        int vid = t + i * 128;
        int r = vid >> 3, c = (vid & 7) * 4;
        aPtr[i]  = A + (long long)(m0 + r) * lda + c;
        aOffS[i] = r * LDA_S + c;
    }
    // B: 4 vec4 per thread, layout depends on mode.
    const TB* bPtr[4]; int bOffS[4]; int bNn[4]; int bKk[4];
    #pragma unroll
    for (int i = 0; i < 4; ++i) {
        int vid = t + i * 128;
        if constexpr (B_TRANS) {            // contiguous along k
            int n = vid >> 3, c = (vid & 7) * 4;
            bPtr[i]  = B + (long long)(n0 + n) * ldb + c;
            bOffS[i] = n * LDB_S + c;
            bNn[i] = 0; bKk[i] = 0;
        } else {                            // contiguous along n, scatter to bT
            int kk = vid >> 4, n = (vid & 15) * 4;
            bPtr[i] = B + (long long)kk * ldb + (n0 + n);
            bNn[i] = n; bKk[i] = kk; bOffS[i] = 0;
        }
    }

    // ---- fragment LDS element offsets (loop-invariant) -------------------
    // A 16x32 f16 layout: lane row = wave*16+lr, VGPR v -> K = 2v + 16*hi
    const int aFragBase = (wave * 16 + lr) * LDA_S + hi * 16;
    // B 32x16 layout: lane col = ns*16+lr, VGPR v -> K = 2v + 16*hi
    const int bFragBase = lr * LDB_S + hi * 16;

    v8f acc[4];
    #pragma unroll
    for (int i = 0; i < 4; ++i)
        #pragma unroll
        for (int j = 0; j < 8; ++j) acc[i][j] = 0.0f;

    V4<TA> av[4]; V4<TB> bv[4];

    // ---- prologue: fetch + stage tile 0 ----------------------------------
    #pragma unroll
    for (int i = 0; i < 4; ++i) av[i] = *(const V4<TA>*)(aPtr[i]);
    #pragma unroll
    for (int i = 0; i < 4; ++i) bv[i] = *(const V4<TB>*)(bPtr[i]);
    #pragma unroll
    for (int i = 0; i < 4; ++i) {
        h4 h = { to_h(av[i].e[0]), to_h(av[i].e[1]), to_h(av[i].e[2]), to_h(av[i].e[3]) };
        *(h4*)&aT[0][aOffS[i]] = h;
    }
    if constexpr (B_TRANS) {
        #pragma unroll
        for (int i = 0; i < 4; ++i) {
            h4 h = { to_h(bv[i].e[0]), to_h(bv[i].e[1]), to_h(bv[i].e[2]), to_h(bv[i].e[3]) };
            *(h4*)&bT[0][bOffS[i]] = h;
        }
    } else {
        #pragma unroll
        for (int i = 0; i < 4; ++i)
            #pragma unroll
            for (int j = 0; j < 4; ++j)
                bT[0][(bNn[i] + j) * LDB_S + bKk[i]] = to_h(bv[i].e[j]);
    }
    __syncthreads();

    const int nk = K / BKK;
    for (int ki = 0; ki < nk; ++ki) {
        const int cur = ki & 1, nxt = cur ^ 1;
        const bool more = (ki + 1 < nk);
        const int k0n = (ki + 1) * BKK;

        // ---- issue next tile's global loads early (hidden behind WMMAs) --
        if (more) {
            #pragma unroll
            for (int i = 0; i < 4; ++i) av[i] = *(const V4<TA>*)(aPtr[i] + k0n);
            if constexpr (B_TRANS) {
                #pragma unroll
                for (int i = 0; i < 4; ++i) bv[i] = *(const V4<TB>*)(bPtr[i] + k0n);
            } else {
                #pragma unroll
                for (int i = 0; i < 4; ++i) bv[i] = *(const V4<TB>*)(bPtr[i] + (long long)k0n * ldb);
            }
        }

        // ---- load ALL fragments first (distinct registers), then 4 WMMAs -
        const _Float16* aB = aT[cur];
        const _Float16* bB = bT[cur];
        Frag af;
        #pragma unroll
        for (int v = 0; v < 8; ++v)
            af.u[v] = *(const unsigned*)&aB[aFragBase + 2 * v];
        Frag bf[4];
        #pragma unroll
        for (int ns = 0; ns < 4; ++ns) {
            const int base = bFragBase + ns * 16 * LDB_S;
            #pragma unroll
            for (int v = 0; v < 8; ++v)
                bf[ns].u[v] = *(const unsigned*)&bB[base + 2 * v];
        }
        #pragma unroll
        for (int ns = 0; ns < 4; ++ns)
            acc[ns] = __builtin_amdgcn_wmma_f32_16x16x32_f16(
                false, af.v, false, bf[ns].v, (short)0, acc[ns], false, false);

        // ---- stage next tile into the other buffer -----------------------
        if (more) {
            #pragma unroll
            for (int i = 0; i < 4; ++i) {
                h4 h = { to_h(av[i].e[0]), to_h(av[i].e[1]), to_h(av[i].e[2]), to_h(av[i].e[3]) };
                *(h4*)&aT[nxt][aOffS[i]] = h;
            }
            if constexpr (B_TRANS) {
                #pragma unroll
                for (int i = 0; i < 4; ++i) {
                    h4 h = { to_h(bv[i].e[0]), to_h(bv[i].e[1]), to_h(bv[i].e[2]), to_h(bv[i].e[3]) };
                    *(h4*)&bT[nxt][bOffS[i]] = h;
                }
            } else {
                #pragma unroll
                for (int i = 0; i < 4; ++i)
                    #pragma unroll
                    for (int j = 0; j < 4; ++j)
                        bT[nxt][(bNn[i] + j) * LDB_S + bKk[i]] = to_h(bv[i].e[j]);
            }
        }
        __syncthreads();
    }

    // ---- epilogue: C/D layout row = r + 8*hi, col = lane&15 --------------
    #pragma unroll
    for (int ns = 0; ns < 4; ++ns) {
        const int ncol = n0 + ns * 16 + lr;
        const float bvs = bias ? bias[ncol] : 0.0f;
        #pragma unroll
        for (int r = 0; r < 8; ++r) {
            const int row = m0 + wave * 16 + r + hi * 8;
            float val = acc[ns][r] + bvs;
            if constexpr (RELU) val = fmaxf(val, 0.0f);
            const long long cidx = offC + (long long)row * ldc + ncol;
            if constexpr (OUT_HALF) Ch[cidx] = (_Float16)val;
            else                    Cf[cidx] = val;
        }
    }
}

// ---------------------------------------------------------------------------
// In-place softmax over rows of length SEQ (one block per row).
// Scores were written unscaled; fold in 1/sqrt(d_k) here.
// ---------------------------------------------------------------------------
__global__ __launch_bounds__(256)
void softmax_kernel(float* __restrict__ attn, float scale)
{
    __shared__ float red[256];
    float* p = attn + (long long)blockIdx.x * SEQ;
    const int t = threadIdx.x;

    float v[8];
    float m = -3.4e38f;
    #pragma unroll
    for (int i = 0; i < 8; ++i) { v[i] = p[t + i * 256] * scale; m = fmaxf(m, v[i]); }
    red[t] = m; __syncthreads();
    for (int s = 128; s > 0; s >>= 1) { if (t < s) red[t] = fmaxf(red[t], red[t + s]); __syncthreads(); }
    m = red[0]; __syncthreads();

    float sum = 0.0f;
    #pragma unroll
    for (int i = 0; i < 8; ++i) { v[i] = __expf(v[i] - m); sum += v[i]; }
    red[t] = sum; __syncthreads();
    for (int s = 128; s > 0; s >>= 1) { if (t < s) red[t] += red[t + s]; __syncthreads(); }
    const float inv = 1.0f / red[0];
    #pragma unroll
    for (int i = 0; i < 8; ++i) p[t + i * 256] = v[i] * inv;
}

// ---------------------------------------------------------------------------
// Fused residual + LayerNorm over rows of length D_MODEL (one block per row):
//   out = LN(in1 + in2) * g + b ;  optionally also emit f16 copy.
// ---------------------------------------------------------------------------
template<bool WRITE_HALF>
__global__ __launch_bounds__(256)
void layernorm_kernel(const float* __restrict__ in1, const float* __restrict__ in2,
                      const float* __restrict__ g, const float* __restrict__ b,
                      float* __restrict__ outf, _Float16* __restrict__ outh)
{
    __shared__ float red[256];
    const long long row = blockIdx.x;
    const float* p1 = in1 + row * D_MODEL;
    const float* p2 = in2 + row * D_MODEL;
    const int t = threadIdx.x;

    float v[4]; float s = 0.0f;
    #pragma unroll
    for (int i = 0; i < 4; ++i) { int c = t + i * 256; v[i] = p1[c] + p2[c]; s += v[i]; }
    red[t] = s; __syncthreads();
    for (int k = 128; k > 0; k >>= 1) { if (t < k) red[t] += red[t + k]; __syncthreads(); }
    const float mu = red[0] * (1.0f / D_MODEL); __syncthreads();

    float sq = 0.0f;
    #pragma unroll
    for (int i = 0; i < 4; ++i) { float d = v[i] - mu; sq += d * d; }
    red[t] = sq; __syncthreads();
    for (int k = 128; k > 0; k >>= 1) { if (t < k) red[t] += red[t + k]; __syncthreads(); }
    const float rstd = rsqrtf(red[0] * (1.0f / D_MODEL) + LN_EPS);

    #pragma unroll
    for (int i = 0; i < 4; ++i) {
        int c = t + i * 256;
        float o = (v[i] - mu) * rstd * g[c] + b[c];
        outf[row * D_MODEL + c] = o;
        if constexpr (WRITE_HALF) outh[row * D_MODEL + c] = (_Float16)o;
    }
}

// ---------------------------------------------------------------------------
extern "C" void kernel_launch(void* const* d_in, const int* in_sizes, int n_in,
                              void* d_out, int out_size, void* d_ws, size_t ws_size,
                              hipStream_t stream)
{
    (void)in_sizes; (void)n_in; (void)out_size; (void)ws_size;

    const float* x     = (const float*)d_in[0];
    const float* Wq    = (const float*)d_in[1];
    const float* bq    = (const float*)d_in[2];
    const float* Wk    = (const float*)d_in[3];
    const float* bk    = (const float*)d_in[4];
    const float* Wv    = (const float*)d_in[5];
    const float* bv    = (const float*)d_in[6];
    const float* Wo    = (const float*)d_in[7];
    const float* bo    = (const float*)d_in[8];
    const float* g1    = (const float*)d_in[9];
    const float* b1    = (const float*)d_in[10];
    const float* W_ff1 = (const float*)d_in[11];
    const float* b_ff1 = (const float*)d_in[12];
    const float* W_ff2 = (const float*)d_in[13];
    const float* b_ff2 = (const float*)d_in[14];
    const float* g2    = (const float*)d_in[15];
    const float* b2    = (const float*)d_in[16];

    const long long BS  = (long long)BSZ * SEQ;        // 4096 token rows
    const long long BSD = BS * D_MODEL;                // 4,194,304
    const long long SS  = (long long)SEQ * SEQ;        // 4,194,304

    // d_out = [ out: BSD f32 | attn: BSZ*H*S*S f32 ]
    float* out_ptr  = (float*)d_out;
    float* attn_ptr = out_ptr + BSD;

    // workspace layout (bytes), with reuse after attention phase:
    char* ws = (char*)d_ws;
    _Float16* Qh   = (_Float16*)(ws + 0);                  //  8 MB
    _Float16* Kh   = (_Float16*)(ws + (8u << 20));         //  8 MB
    _Float16* Vh   = (_Float16*)(ws + (16u << 20));        //  8 MB
    _Float16* ctxh = (_Float16*)(ws + (24u << 20));        //  8 MB
    float*    aout = (float*)   (ws + (32u << 20));        // 16 MB
    float*    hf   = (float*)   (ws + (48u << 20));        // 16 MB
    _Float16* hh   = (_Float16*)(ws + (64u << 20));        //  8 MB
    _Float16* ff1  = (_Float16*)(ws + 0);                  // 32 MB (reuse QKV/ctx)
    float*    ffo  = (float*)   (ws + (32u << 20));        // 16 MB (reuse aout)

    dim3 blk(128);

    // 1-3) QKV projections: x[4096,1024] @ W[1024,1024] + b  -> f16
    {
        dim3 grid(D_MODEL / BN, BS / BM, 1);
        wmma_gemm_kernel<float, float, false, false, true><<<grid, blk, 0, stream>>>(
            x, Wq, bq, nullptr, Qh, (int)BS, D_MODEL, D_MODEL,
            D_MODEL, D_MODEL, D_MODEL, 1, 0, 0, 0, 0, 0, 0);
        wmma_gemm_kernel<float, float, false, false, true><<<grid, blk, 0, stream>>>(
            x, Wk, bk, nullptr, Kh, (int)BS, D_MODEL, D_MODEL,
            D_MODEL, D_MODEL, D_MODEL, 1, 0, 0, 0, 0, 0, 0);
        wmma_gemm_kernel<float, float, false, false, true><<<grid, blk, 0, stream>>>(
            x, Wv, bv, nullptr, Vh, (int)BS, D_MODEL, D_MODEL,
            D_MODEL, D_MODEL, D_MODEL, 1, 0, 0, 0, 0, 0, 0);
    }

    // 4) scores = Q @ K^T per (b,h): [2048,64]x[64,2048] -> attn region (unscaled)
    {
        dim3 grid(SEQ / BN, SEQ / BM, BSZ * N_HEADS);
        wmma_gemm_kernel<_Float16, _Float16, true, false, false><<<grid, blk, 0, stream>>>(
            Qh, Kh, nullptr, attn_ptr, nullptr, SEQ, SEQ, DK,
            D_MODEL, D_MODEL, SEQ, N_HEADS,
            (long long)SEQ * D_MODEL, DK,          // A: b-stride, h-stride
            (long long)SEQ * D_MODEL, DK,          // B: b-stride, h-stride
            (long long)N_HEADS * SS, SS);          // C: b-stride, h-stride
    }

    // 5) softmax in place on attn region (scale 1/sqrt(64) = 0.125)
    softmax_kernel<<<dim3((unsigned)(BSZ * N_HEADS * SEQ)), dim3(256), 0, stream>>>(
        attn_ptr, 0.125f);

    // 6) ctx = attn @ V per (b,h): [2048,2048]x[2048,64] -> f16 [B,S,D] layout
    {
        dim3 grid(DK / BN, SEQ / BM, BSZ * N_HEADS);
        wmma_gemm_kernel<float, _Float16, false, false, true><<<grid, blk, 0, stream>>>(
            attn_ptr, Vh, nullptr, nullptr, ctxh, SEQ, DK, SEQ,
            SEQ, D_MODEL, D_MODEL, N_HEADS,
            (long long)N_HEADS * SS, SS,           // A = attn
            (long long)SEQ * D_MODEL, DK,          // B = V
            (long long)SEQ * D_MODEL, DK);         // C = ctx
    }

    // 7) attn_out = ctx @ Wo + bo  (f32, for residual)
    {
        dim3 grid(D_MODEL / BN, BS / BM, 1);
        wmma_gemm_kernel<_Float16, float, false, false, false><<<grid, blk, 0, stream>>>(
            ctxh, Wo, bo, aout, nullptr, (int)BS, D_MODEL, D_MODEL,
            D_MODEL, D_MODEL, D_MODEL, 1, 0, 0, 0, 0, 0, 0);
    }

    // 8) h = LN(x + attn_out)  -> f32 + f16 copies
    layernorm_kernel<true><<<dim3((unsigned)BS), dim3(256), 0, stream>>>(
        x, aout, g1, b1, hf, hh);

    // 9) ff1 = relu(h @ W_ff1 + b_ff1)  -> f16 [4096,4096]
    {
        dim3 grid(D_FF / BN, BS / BM, 1);
        wmma_gemm_kernel<_Float16, float, false, true, true><<<grid, blk, 0, stream>>>(
            hh, W_ff1, b_ff1, nullptr, ff1, (int)BS, D_FF, D_MODEL,
            D_MODEL, D_FF, D_FF, 1, 0, 0, 0, 0, 0, 0);
    }

    // 10) ff = ff1 @ W_ff2 + b_ff2  -> f32
    {
        dim3 grid(D_MODEL / BN, BS / BM, 1);
        wmma_gemm_kernel<_Float16, float, false, false, false><<<grid, blk, 0, stream>>>(
            ff1, W_ff2, b_ff2, ffo, nullptr, (int)BS, D_MODEL, D_FF,
            D_FF, D_MODEL, D_MODEL, 1, 0, 0, 0, 0, 0, 0);
    }

    // 11) out = LN(h + ff)
    layernorm_kernel<false><<<dim3((unsigned)BS), dim3(256), 0, stream>>>(
        hf, ffo, g2, b2, out_ptr, nullptr);
}
